// UKR_16630113370568
// MI455X (gfx1250) — matrix operand order repaired and checked by reading, not compiled, not run on hardware
//
#include <hip/hip_runtime.h>

typedef __attribute__((ext_vector_type(16))) _Float16 v16h;
typedef __attribute__((ext_vector_type(8)))  float    v8f;
typedef __attribute__((ext_vector_type(8)))  unsigned int v8u;

#define NN 8192
#define DD 512
#define TILE_I 16
#define TILE_J 32                   // K-chunk per WMMA group (f16 path)
#define WAVES_PER_WG 16             // 512 threads
#define TILE_D (WAVES_PER_WG * 16)  // 256 columns of Y per workgroup
#define SEG 8                       // j-segments for the row-sum pass
#define SEG_J (NN / SEG)            // 1024
#define LDS_STRIDE 17               // dword row stride ([kpair][m]), padded vs 16

// exp(-d2/2) == exp2(d2 * -1/(2 ln 2))
#define NEG_HALF_LOG2E (-0.7213475204444817f)

__device__ __forceinline__ float fast_exp_neg_half(float d2) {
    return __builtin_amdgcn_exp2f(d2 * NEG_HALF_LOG2E);   // v_exp_f32
}

// ---------------------------------------------------------------------------
// Pass 1a: partial row sums of K = exp(-d2/2). Grid (NN/256, SEG).
// ---------------------------------------------------------------------------
__global__ __launch_bounds__(256) void rowsum_kernel(const float* __restrict__ Z,
                                                     float* __restrict__ partial) {
    __shared__ float zs[SEG_J * 2];  // 8 KB
    const int tid = threadIdx.x;
    const int seg = blockIdx.y;
    const int jbase = seg * SEG_J;

    for (int t = tid; t < SEG_J * 2; t += 256) zs[t] = Z[jbase * 2 + t];
    __syncthreads();

    const int i = blockIdx.x * 256 + tid;
    const float zi0 = Z[2 * i + 0];
    const float zi1 = Z[2 * i + 1];
    const float sqi = zi0 * zi0 + zi1 * zi1;

    float s = 0.0f;
    for (int j = 0; j < SEG_J; ++j) {
        const float zj0 = zs[2 * j + 0];
        const float zj1 = zs[2 * j + 1];
        float d2 = sqi + zj0 * zj0 + zj1 * zj1 - 2.0f * (zi0 * zj0 + zi1 * zj1);
        d2 = fmaxf(d2, 0.0f);
        s += fast_exp_neg_half(d2);
    }
    partial[seg * NN + i] = s;
}

// ---------------------------------------------------------------------------
// Pass 1b: Sinv[i] = 1 / sum_seg partial[seg][i]
// ---------------------------------------------------------------------------
__global__ __launch_bounds__(256) void reduce_kernel(const float* __restrict__ partial,
                                                     float* __restrict__ Sinv) {
    const int i = blockIdx.x * 256 + threadIdx.x;
    float s = 0.0f;
#pragma unroll
    for (int seg = 0; seg < SEG; ++seg) s += partial[seg * NN + i];
    Sinv[i] = 1.0f / s;
}

// ---------------------------------------------------------------------------
// Pass 2: Y = diag(Sinv) * K @ X, split-precision f16 WMMA.
//   K (on-the-fly exp) and X are split x = hi + lo (f16 each);
//   A*B ~= Ahi*Bhi + Ahi*Blo + Alo*Bhi  (3x v_wmma_f32_16x16x32_f16, f32 acc)
// Workgroup = 512 threads (16 waves), tile = 16 i x 256 d, grid (512, 2).
// Per 32-j chunk every thread produces one K element into padded LDS tiles.
//
// Operand layouts used here (wave32, 16-bit; contiguous k = 16h + s, not AMD's table order):
//   A 16x32: lane l -> m = l%16; dword e holds K pair p = e + (l<16?0:8)
//   B 32x16: lane l -> n = l%16; dword v holds K = (l<16?0:16) + 2v, 2v+1
//   C/D 16x16 f32: lane l -> n = l%16; vgpr v -> m = v + (l<16?0:8)
// ---------------------------------------------------------------------------
__global__ __launch_bounds__(512) void gemm_kernel(const float* __restrict__ X,
                                                   const float* __restrict__ Z,
                                                   const float* __restrict__ Sinv,
                                                   float* __restrict__ Y) {
    __shared__ _Float16 ahi_s[16 * LDS_STRIDE * 2];  // [kpair][m] packed pairs, padded
    __shared__ _Float16 alo_s[16 * LDS_STRIDE * 2];
    __shared__ float sinv_s[TILE_I];

    const int tid  = threadIdx.x;
    const int lane = tid & 31;
    const int wave = tid >> 5;                        // 0..15
    const int i0   = blockIdx.x * TILE_I;
    const int d0   = blockIdx.y * TILE_D + wave * 16;

    const int n     = lane & 15;                      // B/D column
    const int m     = lane & 15;                      // A row
    const int half  = lane >> 4;                      // 0 | 1
    const int aoff  = half * 8;                       // A kpair offset
    const int koffB = half * 16;                      // B K offset

    // K-tile producer role: element (tm, tk), tk in 0..31 -> all 512 threads
    const int tm   = tid & 15;
    const int tk   = tid >> 4;                        // 0..31
    const int tp   = tk >> 1;                         // kpair row
    const int thl  = tk & 1;                          // low/high half of dword
    const int astore = (tp * LDS_STRIDE + tm) * 2 + thl;

    const float zi0 = Z[2 * (i0 + tm) + 0];
    const float zi1 = Z[2 * (i0 + tm) + 1];
    const float sqi = zi0 * zi0 + zi1 * zi1;
    if (tid < TILE_I) sinv_s[tid] = Sinv[i0 + tid];

    const unsigned int* ahi_u = (const unsigned int*)ahi_s;
    const unsigned int* alo_u = (const unsigned int*)alo_s;

    v8f acc = {0.f, 0.f, 0.f, 0.f, 0.f, 0.f, 0.f, 0.f};

    for (int j0 = 0; j0 < NN; j0 += TILE_J) {
        // ---- produce split K tile (one exp per thread) ----
        {
            const float zj0v = Z[2 * (j0 + tk) + 0];
            const float zj1v = Z[2 * (j0 + tk) + 1];
            float d2 = sqi + zj0v * zj0v + zj1v * zj1v
                     - 2.0f * (zi0 * zj0v + zi1 * zj1v);
            d2 = fmaxf(d2, 0.0f);
            const float kf = fast_exp_neg_half(d2);
            const _Float16 kh = (_Float16)kf;
            const _Float16 kl = (_Float16)(kf - (float)kh);
            ahi_s[astore] = kh;
            alo_s[astore] = kl;
        }
        __syncthreads();

        const float* xcol = X + (size_t)(j0 + koffB) * DD + d0 + n;
        if (j0 + TILE_J < NN)
            __builtin_prefetch(xcol + (size_t)TILE_J * DD, 0, 3);

        // ---- A operands from LDS (8 packed-dword loads each) ----
        v8u ahu, alu;
#pragma unroll
        for (int e = 0; e < 8; ++e) {
            const int p = e + aoff;
            ahu[e] = ahi_u[p * LDS_STRIDE + m];
            alu[e] = alo_u[p * LDS_STRIDE + m];
        }
        const v16h ahi = __builtin_bit_cast(v16h, ahu);
        const v16h alo = __builtin_bit_cast(v16h, alu);

        // ---- B operands: load f32 X, split to hi/lo f16 in registers ----
        v16h bhi, blo;
#pragma unroll
        for (int v = 0; v < 8; ++v) {
            const float x0 = xcol[(size_t)(2 * v + 0) * DD];
            const float x1 = xcol[(size_t)(2 * v + 1) * DD];
            const _Float16 h0 = (_Float16)x0;
            const _Float16 h1 = (_Float16)x1;
            bhi[2 * v + 0] = h0;
            bhi[2 * v + 1] = h1;
            blo[2 * v + 0] = (_Float16)(x0 - (float)h0);
            blo[2 * v + 1] = (_Float16)(x1 - (float)h1);
        }

        // ---- 3-term split-precision MAC on the f16 XDL pipe ----
        acc = __builtin_amdgcn_wmma_f32_16x16x32_f16(
            false, ahi, false, bhi, (short)0, acc, false, false);
        acc = __builtin_amdgcn_wmma_f32_16x16x32_f16(
            false, ahi, false, blo, (short)0, acc, false, false);
        acc = __builtin_amdgcn_wmma_f32_16x16x32_f16(
            false, alo, false, bhi, (short)0, acc, false, false);

        __syncthreads();
    }

    // Epilogue: row-normalize and store. acc[v] holds row m = v + half*8.
#pragma unroll
    for (int v = 0; v < 8; ++v) {
        const int row = v + half * 8;
        Y[(size_t)(i0 + row) * DD + d0 + n] = acc[v] * sinv_s[row];
    }
}

// ---------------------------------------------------------------------------
// Pass 3: append Z to the output (reference returns (Y, Z)).
// ---------------------------------------------------------------------------
__global__ __launch_bounds__(256) void copyz_kernel(const float* __restrict__ Z,
                                                    float* __restrict__ out) {
    const int t = blockIdx.x * 256 + threadIdx.x;
    if (t < NN * 2) out[t] = Z[t];
}

extern "C" void kernel_launch(void* const* d_in, const int* in_sizes, int n_in,
                              void* d_out, int out_size, void* d_ws, size_t ws_size,
                              hipStream_t stream) {
    const float* X = (const float*)d_in[0];   // [8192, 512]
    const float* Z = (const float*)d_in[1];   // [8192, 2]
    float* out = (float*)d_out;               // Y (8192*512) then Z (8192*2)

    float* partial = (float*)d_ws;            // SEG * NN floats
    float* Sinv    = partial + SEG * NN;      // NN floats  (total 288 KB of ws)

    rowsum_kernel<<<dim3(NN / 256, SEG), 256, 0, stream>>>(Z, partial);
    reduce_kernel<<<NN / 256, 256, 0, stream>>>(partial, Sinv);
    gemm_kernel<<<dim3(NN / TILE_I, DD / TILE_D), 512, 0, stream>>>(X, Z, Sinv, out);
    copyz_kernel<<<(NN * 2 + 255) / 256, 256, 0, stream>>>(Z, out + (size_t)NN * DD);
}
